// MambaPlusPlusBlock_4793183502529
// MI455X (gfx1250) — matrix operand order repaired and checked by reading, compiled, not run on hardware
//
#include <hip/hip_runtime.h>
#include <hip/hip_bf16.h>

// ---------------------------------------------------------------------------
// MambaPlusPlusBlock for MI455X (gfx1250, wave32, WMMA bf16 16x16x32)
// ---------------------------------------------------------------------------
#define DIM     272
#define DSTATE  64
#define DCONV   4
#define DINNER  544
#define DTRANK  17
#define BATCH   2
#define SEQ     1024
#define NTOK    (BATCH * SEQ)        // 2048
#define KP_IN   288                  // 272 padded to 32
#define NXD     160                  // 145 padded to 16-tile (10 tiles)

static __device__ __constant__ float kLamInit = 0.47071301842f; // 0.8-0.6*exp(-0.6)

typedef __attribute__((ext_vector_type(16))) __bf16 v16bf;
typedef __attribute__((ext_vector_type(8)))  float  v8f;

struct alignas(16) U128 { unsigned int w[4]; };
union Frag { v16bf v; U128 q[2]; };

__device__ __forceinline__ unsigned short f2bf(float f) {
    unsigned int u = __float_as_uint(f);
    u += 0x7FFFu + ((u >> 16) & 1u);          // round-to-nearest-even
    return (unsigned short)(u >> 16);
}

__device__ __forceinline__ float wave_sum(float v) {
    #pragma unroll
    for (int off = 16; off > 0; off >>= 1) v += __shfl_xor(v, off, 32);
    return v;
}

// --------------------------- lambda scalar ---------------------------------
__global__ void k_lambda(const float* __restrict__ lq, float* __restrict__ lam) {
    __shared__ float red[8];
    int tid = threadIdx.x;
    float s = 0.f;
    for (int i = tid; i < DIM; i += 256) s += lq[i];
    s = wave_sum(s);
    if ((tid & 31) == 0) red[tid >> 5] = s;
    __syncthreads();
    if (tid == 0) {
        float t = 0.f;
        #pragma unroll
        for (int i = 0; i < 8; ++i) t += red[i];
        lam[0] = 1.f / (1.f + __expf(-t)) + kLamInit;
    }
}

// ------------------- RMS norm + bf16 cast (K padded) -----------------------
__global__ void k_rmsnorm(const float* __restrict__ x, const float* __restrict__ w,
                          unsigned short* __restrict__ xnbf) {
    int t = blockIdx.x;            // token
    int i = threadIdx.x;           // 0..287
    __shared__ float red[9];
    float v = (i < DIM) ? x[(size_t)t * DIM + i] : 0.f;
    float sq = wave_sum(v * v);
    if ((i & 31) == 0) red[i >> 5] = sq;
    __syncthreads();
    if (i == 0) {
        float s = 0.f;
        #pragma unroll
        for (int k = 0; k < 9; ++k) s += red[k];
        red[0] = rsqrtf(s / (float)DIM + 1e-5f);
    }
    __syncthreads();
    float r = red[0];
    float xn = (i < DIM) ? v * r * w[i] : 0.f;
    xnbf[(size_t)t * KP_IN + i] = f2bf(xn);
}

// -------- pack W (N x K, row-major) into WMMA B-fragment lane layout -------
// Wp[tile=(nt*KT+kt)][lane][e]  where lane<16: B[k=kt*32+e][n=nt*16+lane]
//                                   lane>=16: B[k=kt*32+16+e][n=nt*16+lane-16]
__global__ void k_pack(const float* __restrict__ W, int N, int K, int KT, int total,
                       unsigned short* __restrict__ Wp) {
    int g = blockIdx.x * blockDim.x + threadIdx.x;
    if (g >= total) return;                 // total = Ntiles*KT*32
    int lane = g & 31;
    int tile = g >> 5;
    int kt = tile % KT, nt = tile / KT;
    int n  = nt * 16 + (lane & 15);
    int kb = kt * 32 + (lane >> 4) * 16;
    unsigned short buf[16];
    #pragma unroll
    for (int e = 0; e < 16; ++e) {
        int k = kb + e;
        float w = (n < N && k < K) ? W[(size_t)n * K + k] : 0.f;
        buf[e] = f2bf(w);
    }
    U128* dst = (U128*)(Wp + (size_t)tile * 512 + lane * 16);
    dst[0] = *(U128*)&buf[0];
    dst[1] = *(U128*)&buf[8];
}

// ----------------------- generic WMMA bf16 GEMM ----------------------------
// One wave per 16x16 output tile. A: M x Kp row-major bf16. Wp: packed frags.
// mode 0: plain f32 store. mode 1: softplus(acc + bias[n]) (dt projection).
__global__ void k_gemm(const unsigned short* __restrict__ A,
                       const unsigned short* __restrict__ Wp,
                       float* __restrict__ out,
                       int Kp, int Nld, int mode, const float* __restrict__ bias) {
    int lane = threadIdx.x;
    int tm = blockIdx.x, tn = blockIdx.y;
    int half = lane >> 4, mr = lane & 15;
    v8f acc = {};
    int KT = Kp >> 5;
    const unsigned short* arow = A + (size_t)(tm * 16 + mr) * Kp + half * 16;
    const unsigned short* wp   = Wp + (size_t)tn * KT * 512 + lane * 16;
    for (int kt = 0; kt < KT; ++kt) {
        Frag a, b;
        a.q[0] = *(const U128*)(arow);          // K 0-7   (or 16-23 for half=1)
        a.q[1] = *(const U128*)(arow + 8);      // K 8-15  (or 24-31 for half=1)
        b.q[0] = *(const U128*)(wp);
        b.q[1] = *(const U128*)(wp + 8);
        __builtin_prefetch(wp + 512, 0, 1);     // global_prefetch_b8
        acc = __builtin_amdgcn_wmma_f32_16x16x32_bf16(
                  false, a.v, false, b.v, (short)0, acc, false, false);
        arow += 32;
        wp   += 512;
    }
    int n = tn * 16 + mr;
    int mbase = tm * 16 + (half ? 8 : 0);
    #pragma unroll
    for (int r = 0; r < 8; ++r) {
        float v = acc[r];
        if (mode == 1) {
            v += bias[n];
            v = (v > 20.f) ? v : log1pf(__expf(v));   // softplus
        }
        out[(size_t)(mbase + r) * Nld + n] = v;
    }
}

// ------------------ depthwise causal conv + SiLU (per dir) -----------------
__global__ void k_conv(const float* __restrict__ xz, const float* __restrict__ cw,
                       const float* __restrict__ cb, float* __restrict__ u,
                       unsigned short* __restrict__ ubf, int fwd) {
    int g = blockIdx.x * blockDim.x + threadIdx.x;
    if (g >= NTOK * DINNER) return;
    int c = g % DINNER;
    int t = g / DINNER;
    int l = t % SEQ;
    int base = t - l;                          // b*SEQ
    float acc = cb[c];
    if (fwd) {
        #pragma unroll
        for (int j = 0; j < DCONV; ++j) {
            int ll = l - 3 + j;
            if (ll >= 0) acc += cw[c * DCONV + j] * xz[(size_t)(base + ll) * (2 * DINNER) + c];
        }
    } else {  // mamba(flip(x)) in original index space: anti-causal, taps reversed
        #pragma unroll
        for (int o = 0; o < DCONV; ++o) {
            int ll = l + o;
            if (ll < SEQ) acc += cw[c * DCONV + 3 - o] * xz[(size_t)(base + ll) * (2 * DINNER) + c];
        }
    }
    float s = acc / (1.f + __expf(-acc));      // silu
    u[g] = s;
    ubf[g] = f2bf(s);
}

// ------------------------ dt slice -> bf16 (pad 32) ------------------------
__global__ void k_dtcast(const float* __restrict__ xdbl, unsigned short* __restrict__ dtbf) {
    int g = blockIdx.x * blockDim.x + threadIdx.x;
    if (g >= NTOK * 32) return;
    int k = g & 31;
    int t = g >> 5;
    float v = (k < DTRANK) ? xdbl[(size_t)t * NXD + k] : 0.f;
    dtbf[g] = f2bf(v);
}

// ------------------------- selective scan (per dir) ------------------------
// One wave per (batch, channel). 64 states = 2 per lane. Per-step y reduction
// via shfl_xor tree; exp co-executes on the trans pipe.
__global__ void k_scan(const float* __restrict__ delta, const float* __restrict__ u,
                       const float* __restrict__ xdbl, const float* __restrict__ A_log,
                       float* __restrict__ ys, int fwd) {
    int wave = (blockIdx.x * blockDim.x + threadIdx.x) >> 5;
    int lane = threadIdx.x & 31;
    if (wave >= BATCH * DINNER) return;
    int b = wave / DINNER, d = wave % DINNER;
    float a0 = -__expf(A_log[(size_t)d * DSTATE + lane]);
    float a1 = -__expf(A_log[(size_t)d * DSTATE + 32 + lane]);
    float h0 = 0.f, h1 = 0.f;
    int rbase = b * SEQ;
    for (int it = 0; it < SEQ; ++it) {
        int l = fwd ? it : (SEQ - 1 - it);
        size_t row = (size_t)(rbase + l);
        float dl = delta[row * DINNER + d];
        float uu = u[row * DINNER + d];
        const float* xr = xdbl + row * NXD;
        float du = dl * uu;
        h0 = __expf(dl * a0) * h0 + du * xr[DTRANK + lane];            // B[n]
        h1 = __expf(dl * a1) * h1 + du * xr[DTRANK + 32 + lane];
        float p = h0 * xr[DTRANK + DSTATE + lane]                      // C[n]
                + h1 * xr[DTRANK + DSTATE + 32 + lane];
        p = wave_sum(p);
        if (lane == 0) ys[row * DINNER + d] = p;
    }
}

// --------------------- y = (ys + u*D) * silu(z) -> bf16 --------------------
__global__ void k_gate(const float* __restrict__ ys, const float* __restrict__ u,
                       const float* __restrict__ D, const float* __restrict__ xz,
                       unsigned short* __restrict__ ybf) {
    int g = blockIdx.x * blockDim.x + threadIdx.x;
    if (g >= NTOK * DINNER) return;
    int c = g % DINNER;
    int t = g / DINNER;
    float z = xz[(size_t)t * (2 * DINNER) + DINNER + c];
    float y = (ys[g] + u[g] * D[c]) * (z / (1.f + __expf(-z)));
    ybf[g] = f2bf(y);
}

// ------------- differential combine + RMS + residual epilogue --------------
__global__ void k_final(const float* __restrict__ x, const float* __restrict__ y1,
                        const float* __restrict__ y2, const float* __restrict__ w2,
                        const float* __restrict__ lam, float* __restrict__ out) {
    int t = blockIdx.x, i = threadIdx.x;
    __shared__ float red[9];
    float L = lam[0];
    float diff = 0.f;
    if (i < DIM) diff = y1[(size_t)t * DIM + i] - L * y2[(size_t)t * DIM + i];
    float sq = wave_sum(diff * diff);
    if ((i & 31) == 0) red[i >> 5] = sq;
    __syncthreads();
    if (i == 0) {
        float s = 0.f;
        #pragma unroll
        for (int k = 0; k < 9; ++k) s += red[k];
        red[0] = rsqrtf(s / (float)DIM + 1e-5f);
    }
    __syncthreads();
    if (i < DIM)
        out[(size_t)t * DIM + i] =
            x[(size_t)t * DIM + i] + diff * red[0] * w2[i] * (1.f - kLamInit);
}

// ---------------------------------------------------------------------------
extern "C" void kernel_launch(void* const* d_in, const int* in_sizes, int n_in,
                              void* d_out, int out_size, void* d_ws, size_t ws_size,
                              hipStream_t stream) {
    (void)in_sizes; (void)n_in; (void)out_size; (void)ws_size;
    const float* x   = (const float*)d_in[0];
    const float* lq  = (const float*)d_in[1];
    const float* n1w = (const float*)d_in[2];
    const float* n2w = (const float*)d_in[3];
    // per-dir params: 0 in_w, 1 conv_w, 2 conv_b, 3 xproj_w, 4 dtproj_w,
    //                 5 dtproj_b, 6 A_log, 7 D, 8 out_w
    const float* P[2][9];
    for (int dir = 0; dir < 2; ++dir)
        for (int k = 0; k < 9; ++k)
            P[dir][k] = (const float*)d_in[4 + dir * 9 + k];

    // --------------------------- workspace layout --------------------------
    char* ws = (char*)d_ws;
    size_t off = 0;
    auto alloc = [&](size_t bytes) {
        void* p = ws + off;
        off = (off + bytes + 255) & ~(size_t)255;
        return p;
    };
    float*          lam   = (float*)         alloc(4);
    unsigned short* xnbf  = (unsigned short*)alloc((size_t)NTOK * KP_IN * 2);
    unsigned short* inpk[2], *xppk[2], *dtpk[2], *outpk[2];
    for (int d = 0; d < 2; ++d) {
        inpk[d]  = (unsigned short*)alloc((size_t)68 * 9  * 512 * 2);  // N=1088,K=288
        xppk[d]  = (unsigned short*)alloc((size_t)10 * 17 * 512 * 2);  // N=160, K=544
        dtpk[d]  = (unsigned short*)alloc((size_t)34 * 1  * 512 * 2);  // N=544, K=32
        outpk[d] = (unsigned short*)alloc((size_t)17 * 17 * 512 * 2);  // N=272, K=544
    }
    float*          xz    = (float*)         alloc((size_t)NTOK * 2 * DINNER * 4);
    float*          u     = (float*)         alloc((size_t)NTOK * DINNER * 4);
    unsigned short* ubf   = (unsigned short*)alloc((size_t)NTOK * DINNER * 2);
    float*          xdbl  = (float*)         alloc((size_t)NTOK * NXD * 4);
    unsigned short* dtbf  = (unsigned short*)alloc((size_t)NTOK * 32 * 2);
    float*          delta = (float*)         alloc((size_t)NTOK * DINNER * 4);
    float*          yscan = (float*)         alloc((size_t)NTOK * DINNER * 4);
    unsigned short* ybf   = (unsigned short*)alloc((size_t)NTOK * DINNER * 2);
    float*          y12[2];
    y12[0] = (float*)alloc((size_t)NTOK * DIM * 4);
    y12[1] = (float*)alloc((size_t)NTOK * DIM * 4);

    // --------------------------- prologue ----------------------------------
    k_lambda <<<1, 256, 0, stream>>>(lq, lam);
    k_rmsnorm<<<NTOK, 288, 0, stream>>>(x, n1w, xnbf);
    for (int d = 0; d < 2; ++d) {
        int t;
        t = 68 * 9 * 32;  k_pack<<<(t + 255) / 256, 256, 0, stream>>>(P[d][0], 2 * DINNER, DIM,    9,  t, inpk[d]);
        t = 10 * 17 * 32; k_pack<<<(t + 255) / 256, 256, 0, stream>>>(P[d][3], DTRANK + 2 * DSTATE, DINNER, 17, t, xppk[d]);
        t = 34 * 1 * 32;  k_pack<<<(t + 255) / 256, 256, 0, stream>>>(P[d][4], DINNER, DTRANK, 1,  t, dtpk[d]);
        t = 17 * 17 * 32; k_pack<<<(t + 255) / 256, 256, 0, stream>>>(P[d][8], DIM,    DINNER, 17, t, outpk[d]);
    }

    // ----------------------- per-direction pipeline ------------------------
    const int ew = 256;
    const int nElemCD = (NTOK * DINNER + ew - 1) / ew;    // 4352 blocks
    for (int dir = 0; dir < 2; ++dir) {
        int fwd = (dir == 0);
        // in-projection: 2048 x 1088 = xn(2048x288bf) @ in_w^T
        k_gemm<<<dim3(NTOK / 16, 68), 32, 0, stream>>>(xnbf, inpk[dir], xz,
                                                       KP_IN, 2 * DINNER, 0, nullptr);
        // depthwise conv + SiLU (direction-aware)
        k_conv<<<nElemCD, ew, 0, stream>>>(xz, P[dir][1], P[dir][2], u, ubf, fwd);
        // x-projection: 2048 x 160
        k_gemm<<<dim3(NTOK / 16, 10), 32, 0, stream>>>(ubf, xppk[dir], xdbl,
                                                       DINNER, NXD, 0, nullptr);
        // dt -> bf16 (pad K to 32)
        k_dtcast<<<(NTOK * 32 + ew - 1) / ew, ew, 0, stream>>>(xdbl, dtbf);
        // dt-projection + softplus: delta 2048 x 544
        k_gemm<<<dim3(NTOK / 16, 34), 32, 0, stream>>>(dtbf, dtpk[dir], delta,
                                                       32, DINNER, 1, P[dir][5]);
        // selective scan: 1088 waves (one per batch*channel), 8 waves/block
        k_scan<<<(BATCH * DINNER) / 8, 256, 0, stream>>>(delta, u, xdbl, P[dir][6],
                                                         yscan, fwd);
        // gate: (ys + u*D) * silu(z) -> bf16
        k_gate<<<nElemCD, ew, 0, stream>>>(yscan, u, P[dir][7], xz, ybf);
        // out-projection: 2048 x 272
        k_gemm<<<dim3(NTOK / 16, 17), 32, 0, stream>>>(ybf, outpk[dir], y12[dir],
                                                       DINNER, DIM, 0, nullptr);
    }

    // --------------------------- epilogue ----------------------------------
    k_final<<<NTOK, 288, 0, stream>>>(x, y12[0], y12[1], n2w, lam, (float*)d_out);
}
